// Embedding_28123445854678
// MI455X (gfx1250) — compile-verified, matrix operand order checked by reading, not run on hardware
//
#include <hip/hip_runtime.h>

typedef __attribute__((ext_vector_type(16))) _Float16 v16h;
typedef __attribute__((ext_vector_type(8)))  float    v8f;

// 16-bit A/B WMMA fragment: two contiguous 8-half (16B) chunks per lane.
// lanes 0-15: K = [0..7] and [16..23]; lanes 16-31: K = [8..15] and [24..31].
union AV {
  v16h     v;
  _Float16 h[16];
  uint4    q[2];
};

#define S3f  1.7320508075688772f
#define S6f  2.449489742783178f
#define S10f 3.1622776601683795f
#define S15f 3.872983346207417f
#define PIf  3.14159265358979323846f
#define RBF_START 0.006737946999085467f   // exp(-5)

// ---------------------------------------------------------------------------
// Prep: convert weights to f16, transposed to N-major so B-fragments load as
// two contiguous 16B chunks per lane.
// ---------------------------------------------------------------------------
__global__ void prep_weights(const float* __restrict__ ndp, const float* __restrict__ erp,
                             const float* __restrict__ l1w, const float* __restrict__ l2w,
                             _Float16* __restrict__ bt_ndp, _Float16* __restrict__ bt_erp,
                             _Float16* __restrict__ bt1, _Float16* __restrict__ bt2) {
  unsigned tid = blockIdx.x * blockDim.x + threadIdx.x;
  unsigned stride = gridDim.x * blockDim.x;
  for (unsigned i = tid; i < 128u * 32u; i += stride) {   // [32,128] -> [128][32]
    unsigned c = i >> 5, k = i & 31;
    bt_ndp[c * 32u + k] = (_Float16)ndp[k * 128u + c];
    bt_erp[c * 32u + k] = (_Float16)erp[k * 128u + c];
  }
  for (unsigned i = tid; i < 128u * 256u; i += stride) {  // [256,128] -> [128][256]
    unsigned c = i >> 8, k = i & 255;
    bt1[c * 256u + k] = (_Float16)l1w[k * 128u + c];
  }
  for (unsigned i = tid; i < 128u * 128u; i += stride) {  // [128,128] -> [128][128]
    unsigned c = i >> 7, k = i & 127;
    bt2[c * 128u + k] = (_Float16)l2w[k * 128u + c];
  }
}

// ---------------------------------------------------------------------------
// Kernel A: per-edge geometry + rbf, then filt = (rbf@Wndp + b)*cut via WMMA,
// msg = filt * a_nbr[z[n_j]], atomic scatter into m_i.
// One wave handles 16 edges. 8 waves / block, E % 128 == 0.
// ---------------------------------------------------------------------------
__global__ __launch_bounds__(256) void edge_msg_kernel(
    const float* __restrict__ p, const int* __restrict__ ei,
    const int* __restrict__ z, const float* __restrict__ a_nbr,
    const _Float16* __restrict__ bt_ndp, const float* __restrict__ ndp_b,
    _Float16* __restrict__ rbf16, float* __restrict__ m_i,
    float* __restrict__ out_r, float* __restrict__ out_sh, int E) {
  __shared__ _Float16 s_rbf[8][16 * 32];
  __shared__ float s_cut[8][16];
  __shared__ unsigned s_zj[8][16];
  __shared__ unsigned s_ni[8][16];

  const unsigned wave = threadIdx.x >> 5;
  const unsigned lane = threadIdx.x & 31;
  const unsigned m    = lane & 15;
  const unsigned half = lane >> 4;
  const unsigned e0   = (blockIdx.x * 8u + wave) * 16u;
  const unsigned e    = e0 + m;

  // ---- geometry (both halves compute redundantly; half 0 writes) ----
  float x = p[e * 3u + 0u], y = p[e * 3u + 1u], zc = p[e * 3u + 2u];
  float r = sqrtf(x * x + y * y + zc * zc);
  float inv = 1.0f / r;
  float ux = x * inv, uy = y * inv, uz = zc * inv;
  float cut = (r < 5.0f) ? 0.5f * (__cosf(PIf * r * 0.2f) + 1.0f) : 0.0f;
  float ear = __expf(-r);  // alpha = 5/CUTOFF = 1

  if (half == 0u) {
    __builtin_nontemporal_store(r, &out_r[e]);
    float x2 = ux * ux, y2 = uy * uy, z2 = uz * uz;
    float sh[15];
    sh[0] = ux; sh[1] = uy; sh[2] = uz;
    sh[3] = S3f * ux * uz;
    sh[4] = S3f * ux * uy;
    sh[5] = y2 - 0.5f * (x2 + z2);
    sh[6] = S3f * uy * uz;
    sh[7] = 0.5f * S3f * (z2 - x2);
    sh[8]  = 0.25f * S10f * ux * (3.0f * z2 - x2);
    sh[9]  = S15f * ux * uy * uz;
    sh[10] = 0.25f * S6f * ux * (4.0f * y2 - x2 - z2);
    sh[11] = 0.5f * uy * (2.0f * y2 - 3.0f * (x2 + z2));
    sh[12] = 0.25f * S6f * uz * (4.0f * y2 - x2 - z2);
    sh[13] = 0.5f * S15f * uy * (z2 - x2);
    sh[14] = 0.25f * S10f * uz * (z2 - 3.0f * x2);
#pragma unroll
    for (unsigned i = 0; i < 15u; i++)
      __builtin_nontemporal_store(sh[i], &out_sh[e * 15u + i]);
    s_cut[wave][m] = cut;
    unsigned nj = (unsigned)ei[e];
    s_zj[wave][m] = (unsigned)z[nj];
    s_ni[wave][m] = (unsigned)ei[(unsigned)E + e];
  }

  // ---- rbf: each lane computes 16 of its edge's 32 gaussians ----
  const float delta = (1.0f - RBF_START) / 31.0f;
  const float gamma = 1.0f / (4.0f * delta * delta);
  _Float16 tmp[16];
#pragma unroll
  for (unsigned kk = 0; kk < 16u; kk++) {
    unsigned k = half * 16u + kk;
    float c = RBF_START + delta * (float)k;
    float d = ear - c;
    tmp[kk] = (_Float16)(__expf(-gamma * d * d) * cut);
  }
  // 2x16B packed stores to LDS and to global workspace
  *(uint4*)&s_rbf[wave][m * 32u + half * 16u]      = *(const uint4*)&tmp[0];
  *(uint4*)&s_rbf[wave][m * 32u + half * 16u + 8u] = *(const uint4*)&tmp[8];
  *(uint4*)&rbf16[e * 32u + half * 16u]            = *(const uint4*)&tmp[0];
  *(uint4*)&rbf16[e * 32u + half * 16u + 8u]       = *(const uint4*)&tmp[8];
  __syncthreads();

  // ---- WMMA: filt tile = rbf[16x32] @ Wndp[32x128] ----
  const unsigned k0 = half ? 8u : 0u;
  AV A;
  A.q[0] = *(const uint4*)&s_rbf[wave][m * 32u + k0];
  A.q[1] = *(const uint4*)&s_rbf[wave][m * 32u + k0 + 16u];
  const unsigned mD = half << 3;
#pragma unroll
  for (unsigned nb = 0; nb < 8u; nb++) {
    const unsigned c = nb * 16u + m;
    AV B;
    B.q[0] = *(const uint4*)&bt_ndp[c * 32u + k0];
    B.q[1] = *(const uint4*)&bt_ndp[c * 32u + k0 + 16u];
    v8f acc = {};
    acc = __builtin_amdgcn_wmma_f32_16x16x32_f16(false, A.v, false, B.v,
                                                 (short)0, acc, false, false);
    const float bias = ndp_b[c];
#pragma unroll
    for (unsigned j = 0; j < 8u; j++) {
      const unsigned row = j + mD;                 // D row -> edge e0+row
      const float val = (acc[j] + bias) * s_cut[wave][row] *
                        a_nbr[s_zj[wave][row] * 128u + c];
      atomicAdd(&m_i[s_ni[wave][row] * 128u + c], val);
    }
  }
}

// ---------------------------------------------------------------------------
// Kernel B: per-node  h = silu(LN([a_na[z], m_i] @ lin1 + b1)) @ lin2 + b2
// One wave per 16 nodes (single-wave workgroup). N % 16 == 0.
// ---------------------------------------------------------------------------
__global__ __launch_bounds__(32) void node_kernel(
    const int* __restrict__ z, const float* __restrict__ a_na,
    const float* __restrict__ m_i,
    const _Float16* __restrict__ bt1, const float* __restrict__ lin1_b,
    const float* __restrict__ ln_g, const float* __restrict__ ln_b,
    const _Float16* __restrict__ bt2, const float* __restrict__ lin2_b,
    float* __restrict__ h_out, int N) {
  __shared__ unsigned s_z[16];
  __shared__ _Float16 s_h[16 * 136];   // 16 rows x 128 cols, padded (272B row)

  const unsigned lane = threadIdx.x;
  const unsigned m = lane & 15, half = lane >> 4;
  const unsigned v0 = blockIdx.x * 16u;
  if (half == 0u) s_z[m] = (unsigned)z[v0 + m];
  __syncthreads();

  const unsigned k0 = half ? 8u : 0u;
  const unsigned vrow = v0 + m;

  // ---- lin1: [16 x 256] @ [256 x 128], 8 K-steps x 8 N-blocks ----
  v8f acc[8] = {};
  for (unsigned ks = 0; ks < 8u; ks++) {
    AV A;
    const unsigned base = ks * 32u;
#pragma unroll
    for (unsigned chunk = 0; chunk < 2u; chunk++) {
      const unsigned k = base + k0 + chunk * 16u;  // 8-wide chunk, never straddles 128
      const float* src = (k < 128u) ? (a_na + s_z[m] * 128u + k)
                                    : (m_i + vrow * 128u + (k - 128u));
#pragma unroll
      for (unsigned i = 0; i < 8u; i++) A.h[chunk * 8u + i] = (_Float16)src[i];
    }
#pragma unroll
    for (unsigned nb = 0; nb < 8u; nb++) {
      const unsigned c = nb * 16u + m;
      AV B;
      B.q[0] = *(const uint4*)&bt1[c * 256u + base + k0];
      B.q[1] = *(const uint4*)&bt1[c * 256u + base + k0 + 16u];
      acc[nb] = __builtin_amdgcn_wmma_f32_16x16x32_f16(false, A.v, false, B.v,
                                                       (short)0, acc[nb], false, false);
    }
  }

  // ---- bias + LayerNorm + SiLU, transpose through LDS ----
  float bias1[8], lg[8], lb[8];
#pragma unroll
  for (unsigned nb = 0; nb < 8u; nb++) {
    const unsigned c = nb * 16u + m;
    bias1[nb] = lin1_b[c]; lg[nb] = ln_g[c]; lb[nb] = ln_b[c];
  }
#pragma unroll
  for (unsigned j = 0; j < 8u; j++) {
    float xv[8], xs = 0.0f, x2s = 0.0f;
#pragma unroll
    for (unsigned nb = 0; nb < 8u; nb++) {
      const float xx = acc[nb][j] + bias1[nb];
      xv[nb] = xx; xs += xx; x2s += xx * xx;
    }
    // row sum across the 16 lanes of this half-wave
#pragma unroll
    for (int s = 1; s < 16; s <<= 1) {
      xs  += __shfl_xor(xs,  s, 16);
      x2s += __shfl_xor(x2s, s, 16);
    }
    const float mean = xs * (1.0f / 128.0f);
    const float var  = x2s * (1.0f / 128.0f) - mean * mean;
    const float rstd = rsqrtf(var + 1e-5f);
    const unsigned row = j + (half << 3);
#pragma unroll
    for (unsigned nb = 0; nb < 8u; nb++) {
      const float yn = (xv[nb] - mean) * rstd * lg[nb] + lb[nb];
      const float sil = yn / (1.0f + __expf(-yn));
      s_h[row * 136u + nb * 16u + m] = (_Float16)sil;
    }
  }
  __syncthreads();

  // ---- lin2: [16 x 128] @ [128 x 128], 4 K-steps x 8 N-blocks ----
  v8f acc2[8] = {};
  for (unsigned ks = 0; ks < 4u; ks++) {
    AV A;
    const _Float16* rowp = &s_h[m * 136u];
    A.q[0] = *(const uint4*)&rowp[ks * 32u + k0];
    A.q[1] = *(const uint4*)&rowp[ks * 32u + k0 + 16u];
#pragma unroll
    for (unsigned nb = 0; nb < 8u; nb++) {
      const unsigned c = nb * 16u + m;
      AV B;
      B.q[0] = *(const uint4*)&bt2[c * 128u + ks * 32u + k0];
      B.q[1] = *(const uint4*)&bt2[c * 128u + ks * 32u + k0 + 16u];
      acc2[nb] = __builtin_amdgcn_wmma_f32_16x16x32_f16(false, A.v, false, B.v,
                                                        (short)0, acc2[nb], false, false);
    }
  }
#pragma unroll
  for (unsigned nb = 0; nb < 8u; nb++) {
    const unsigned c = nb * 16u + m;
    const float b2 = lin2_b[c];
#pragma unroll
    for (unsigned j = 0; j < 8u; j++) {
      const unsigned row = j + (half << 3);
      h_out[(v0 + row) * 128u + c] = acc2[nb][j] + b2;
    }
  }
}

// ---------------------------------------------------------------------------
// Kernel C: t_ij = (h[n_i] + h[n_j]) * (rbf @ Werp + b). Wave per 16 edges.
// h (20 MB) and rbf16 (41 MB) stay L2-resident; t_ij streams out non-temporal.
// ---------------------------------------------------------------------------
__global__ __launch_bounds__(256) void edge_out_kernel(
    const int* __restrict__ ei, const _Float16* __restrict__ rbf16,
    const _Float16* __restrict__ bt_erp, const float* __restrict__ erp_b,
    const float* __restrict__ h, float* __restrict__ t_out, int E) {
  __shared__ unsigned s_ni[8][16], s_nj[8][16];
  const unsigned wave = threadIdx.x >> 5;
  const unsigned lane = threadIdx.x & 31;
  const unsigned m = lane & 15, half = lane >> 4;
  const unsigned e0 = (blockIdx.x * 8u + wave) * 16u;
  if (half == 0u) {
    const unsigned e = e0 + m;
    s_nj[wave][m] = (unsigned)ei[e];
    s_ni[wave][m] = (unsigned)ei[(unsigned)E + e];
  }
  __syncthreads();

  const unsigned k0 = half ? 8u : 0u;
  AV A;
  const unsigned arow = (e0 + m) * 32u;
  A.q[0] = *(const uint4*)&rbf16[arow + k0];
  A.q[1] = *(const uint4*)&rbf16[arow + k0 + 16u];
  const unsigned mD = half << 3;
#pragma unroll
  for (unsigned nb = 0; nb < 8u; nb++) {
    const unsigned c = nb * 16u + m;
    AV B;
    B.q[0] = *(const uint4*)&bt_erp[c * 32u + k0];
    B.q[1] = *(const uint4*)&bt_erp[c * 32u + k0 + 16u];
    v8f acc = {};
    acc = __builtin_amdgcn_wmma_f32_16x16x32_f16(false, A.v, false, B.v,
                                                 (short)0, acc, false, false);
    const float bias = erp_b[c];
#pragma unroll
    for (unsigned j = 0; j < 8u; j++) {
      const unsigned row = j + mD;
      const unsigned e = e0 + row;
      const float hv = h[s_ni[wave][row] * 128u + c] +
                       h[s_nj[wave][row] * 128u + c];
      __builtin_nontemporal_store(hv * (acc[j] + bias), &t_out[e * 128u + c]);
    }
  }
}

// ---------------------------------------------------------------------------
extern "C" void kernel_launch(void* const* d_in, const int* in_sizes, int n_in,
                              void* d_out, int out_size, void* d_ws, size_t ws_size,
                              hipStream_t stream) {
  (void)n_in; (void)out_size; (void)ws_size;
  const int*   z       = (const int*)d_in[0];
  const float* p       = (const float*)d_in[1];
  const int*   ei      = (const int*)d_in[2];
  const float* a_nbr   = (const float*)d_in[3];
  const float* a_na    = (const float*)d_in[4];
  const float* w_ndp_w = (const float*)d_in[5];
  const float* w_ndp_b = (const float*)d_in[6];
  const float* lin1_w  = (const float*)d_in[7];
  const float* lin1_b  = (const float*)d_in[8];
  const float* ln_g    = (const float*)d_in[9];
  const float* ln_b    = (const float*)d_in[10];
  const float* lin2_w  = (const float*)d_in[11];
  const float* lin2_b  = (const float*)d_in[12];
  const float* w_erp_w = (const float*)d_in[13];
  const float* w_erp_b = (const float*)d_in[14];

  const int N = in_sizes[0];        // z: (N,1)
  const int E = in_sizes[1] / 3;    // p: (E,3)

  float* out    = (float*)d_out;
  float* out_r  = out;                                  // (E,1)
  float* out_sh = out + (long)E;                        // (E,15)
  float* out_h  = out + (long)E * 16;                   // (N,128)
  float* out_t  = out + (long)E * 16 + (long)N * 128;   // (E,128)

  char* ws = (char*)d_ws;
  _Float16* rbf16 = (_Float16*)ws;                          // E*32 f16
  float* m_i = (float*)(ws + (size_t)E * 64);               // N*128 f32
  size_t woff = (size_t)E * 64 + (size_t)N * 512;
  _Float16* bt_ndp = (_Float16*)(ws + woff);
  _Float16* bt_erp = bt_ndp + 128 * 32;
  _Float16* bt1    = bt_erp + 128 * 32;
  _Float16* bt2    = bt1 + 128 * 256;

  hipMemsetAsync(m_i, 0, (size_t)N * 512, stream);
  prep_weights<<<32, 256, 0, stream>>>(w_ndp_w, w_erp_w, lin1_w, lin2_w,
                                       bt_ndp, bt_erp, bt1, bt2);
  edge_msg_kernel<<<E / 128, 256, 0, stream>>>(p, ei, z, a_nbr, bt_ndp, w_ndp_b,
                                               rbf16, m_i, out_r, out_sh, E);
  node_kernel<<<N / 16, 32, 0, stream>>>(z, a_na, m_i, bt1, lin1_b, ln_g, ln_b,
                                         bt2, lin2_b, out_h, N);
  edge_out_kernel<<<E / 128, 256, 0, stream>>>(ei, rbf16, bt_erp, w_erp_b,
                                               out_h, out_t, E);
}
